// Encoder_63024350101963
// MI455X (gfx1250) — matrix operand order repaired and checked by reading, not compiled, not run on hardware
//
#include <hip/hip_runtime.h>
#include <hip/hip_bf16.h>

typedef __attribute__((ext_vector_type(16))) _Float16 v16h;
typedef __attribute__((ext_vector_type(8)))  _Float16 v8h;
typedef __attribute__((ext_vector_type(4)))  _Float16 v4h;
typedef __attribute__((ext_vector_type(8)))  float    v8f;

#define B_TOT   2048
#define TM1     31
#define DD      128
#define HH      128
#define G4      512        // 4*H
#define BM      16         // batch rows per block
#define NTHREADS 512       // 16 waves: 0-7 -> gates i,f ; 8-15 -> gates g,o

__device__ __forceinline__ float sigm(float x) { return 1.0f / (1.0f + __expf(-x)); }

// A-fragment (16x32 f16, M x K): lane m=lane&15; hi=lane>>4 selects K halves.
__device__ __forceinline__ v16h loadA(const _Float16* base, int m, int hi, int kb) {
    const v8h a0 = *(const v8h*)(base + m * DD + kb + hi * 8);
    const v8h a1 = *(const v8h*)(base + m * DD + kb + 16 + hi * 8);
    return __builtin_shufflevector(a0, a1, 0,1,2,3,4,5,6,7,8,9,10,11,12,13,14,15);
}

// B-fragment (32x16 f16, K x N) where B[k][n] = W[nglob][k], W row-major [512][128].
__device__ __forceinline__ v16h loadB(const _Float16* w, int nglob, int hi, int kb) {
    const _Float16* p = w + nglob * DD + kb + hi * 8;
    const v8h b0 = *(const v8h*)(p);
    const v8h b1 = *(const v8h*)(p + 16);
    return __builtin_shufflevector(b0, b1, 0,1,2,3,4,5,6,7,8,9,10,11,12,13,14,15);
}

__global__ __launch_bounds__(NTHREADS, 1)
void encoder_lstm_attn_kernel(const float* __restrict__ x,     // (B,31,128)
                              const float* __restrict__ Wih,   // (512,128)
                              const float* __restrict__ Whh,   // (512,128)
                              const float* __restrict__ bih,   // (512)
                              const float* __restrict__ bhh,   // (512)
                              const float* __restrict__ attnw, // (287)
                              const float* __restrict__ attnb, // (1)
                              float* __restrict__ out_w,       // (B,31,128)
                              float* __restrict__ out_h)       // (B,31,128)
{
    __shared__ __align__(16) _Float16 sWih[G4 * DD];   // 128 KB
    __shared__ __align__(16) _Float16 sWhh[G4 * DD];   // 128 KB
    __shared__ __align__(16) _Float16 sAwx[BM * DD];   // 4 KB
    __shared__ __align__(16) _Float16 sAh [BM * DD];   // 4 KB
    __shared__ __align__(16) float    sC  [BM * HH];   // 8 KB
    __shared__ __align__(16) float    sG  [BM * HH];   // 8 KB  (g-gate exchange)
    __shared__ __align__(16) float    sO  [BM * HH];   // 8 KB  (o-gate exchange)
    __shared__ __align__(16) float    sAtt[2 * HH + TM1 + 1];

    const int tid  = threadIdx.x;
    const int wv   = tid >> 5;          // wave id 0..15
    const int lane = tid & 31;
    const int nloc = lane & 15;
    const int hi   = lane >> 4;
    const int blk  = blockIdx.x;

    // ---- one-time setup ----
    for (int i = tid; i < G4 * DD; i += NTHREADS) {
        sWih[i] = (_Float16)Wih[i];
        sWhh[i] = (_Float16)Whh[i];
    }
    for (int i = tid; i < 2 * HH + TM1; i += NTHREADS) sAtt[i] = attnw[i];
    for (int i = tid; i < BM * DD; i += NTHREADS) { sAh[i] = (_Float16)0.f; sC[i] = 0.f; }
    __syncthreads();

    // wave -> (h-col group, gate pair): waves 0-7 => gates 0,1 (i,f); 8-15 => gates 2,3 (g,o)
    const int w7    = wv & 7;
    const int hcolw = w7 * 16 + nloc;
    const int g0    = (wv >> 3) * 2;
    const int g1    = g0 + 1;

    // resident W_hh fragments: 2 gates x 4 k-steps x 8 VGPRs = 64 VGPRs
    v16h bhh0[4], bhh1[4];
#pragma unroll
    for (int kk = 0; kk < 4; kk++) {
        bhh0[kk] = loadB(sWhh, g0 * HH + hcolw, hi, kk * 32);
        bhh1[kk] = loadB(sWhh, g1 * HH + hcolw, hi, kk * 32);
    }

    // ---- row-parallel mapping: one wave per batch row, 4 columns per lane ----
    const int r  = tid >> 5;            // row == wave id
    const int q  = lane;                // 32 lanes per row
    const int d0 = q * 4;
    const long brow = (long)blk * BM + r;
    const float* xrow = x + brow * (TM1 * DD);

    // x_score (registers, 4 per lane)
    float xs[4] = {0.f, 0.f, 0.f, 0.f};
    for (int t = 0; t < TM1; t++) {
        const float wxt = sAtt[2 * HH + t];
        const float4 xa = *(const float4*)(xrow + t * DD + d0);
        xs[0] = fmaf(xa.x, wxt, xs[0]); xs[1] = fmaf(xa.y, wxt, xs[1]);
        xs[2] = fmaf(xa.z, wxt, xs[2]); xs[3] = fmaf(xa.w, wxt, xs[3]);
    }
    float whr[4], wcr[4];
#pragma unroll
    for (int j = 0; j < 4; j++) { whr[j] = sAtt[d0 + j]; wcr[j] = sAtt[HH + d0 + j]; }
    const float ab = attnb[0];

    const float bia = bih[g0 * HH + hcolw] + bhh[g0 * HH + hcolw];
    const float bib = bih[g1 * HH + hcolw] + bhh[g1 * HH + hcolw];

    // ---- recurrent loop over 31 timesteps ----
    for (int t = 0; t < TM1; t++) {
        // phase 1: rowdot = h.w_h + c.w_c + attn_b  (full-wave reduction, one row per wave)
        const v4h  hv4 = *(const v4h*)(sAh + r * DD + d0);
        const float4 c4 = *(const float4*)(sC + r * HH + d0);
        float part = fmaf((float)hv4[0], whr[0], c4.x * wcr[0]);
        part = fmaf((float)hv4[1], whr[1], fmaf(c4.y, wcr[1], part));
        part = fmaf((float)hv4[2], whr[2], fmaf(c4.z, wcr[2], part));
        part = fmaf((float)hv4[3], whr[3], fmaf(c4.w, wcr[3], part));
#pragma unroll
        for (int m = 1; m < 32; m <<= 1) part += __shfl_xor(part, m, 32);
        const float rowdot = part + ab;

        // phase 2: softmax over this row's 128 columns
        float sc[4];
        float rmax = -3.0e38f;
#pragma unroll
        for (int j = 0; j < 4; j++) { sc[j] = rowdot + xs[j]; rmax = fmaxf(rmax, sc[j]); }
#pragma unroll
        for (int m = 1; m < 32; m <<= 1) rmax = fmaxf(rmax, __shfl_xor(rmax, m, 32));
        float rsum = 0.f;
#pragma unroll
        for (int j = 0; j < 4; j++) { sc[j] = __expf(sc[j] - rmax); rsum += sc[j]; }
#pragma unroll
        for (int m = 1; m < 32; m <<= 1) rsum += __shfl_xor(rsum, m, 32);
        const float inv = 1.0f / rsum;

        // phase 3: wx = attn * x_t -> global (b128) + LDS A-matrix (b64)
        const float4 xt4 = *(const float4*)(xrow + t * DD + d0);
        if (t + 1 < TM1) __builtin_prefetch(xrow + (t + 1) * DD + d0, 0, 1);
        float wx0 = sc[0] * inv * xt4.x, wx1 = sc[1] * inv * xt4.y;
        float wx2 = sc[2] * inv * xt4.z, wx3 = sc[3] * inv * xt4.w;
        *(float4*)(out_w + brow * (TM1 * DD) + t * DD + d0) = make_float4(wx0, wx1, wx2, wx3);
        v4h wxh; wxh[0] = (_Float16)wx0; wxh[1] = (_Float16)wx1;
                 wxh[2] = (_Float16)wx2; wxh[3] = (_Float16)wx3;
        *(v4h*)(sAwx + r * DD + d0) = wxh;
        __syncthreads();

        // phase 4: gates = wx @ Wih^T + h @ Whh^T + bias  (WMMA, software-pipelined)
        v8f acc0, acc1;
#pragma unroll
        for (int e = 0; e < 8; e++) { acc0[e] = bia; acc1[e] = bib; }
        v16h cw0 = loadB(sWih, g0 * HH + hcolw, hi, 0);
        v16h cw1 = loadB(sWih, g1 * HH + hcolw, hi, 0);
        v16h cah = loadA(sAh,  nloc, hi, 0);
        v16h caw = loadA(sAwx, nloc, hi, 0);
#pragma unroll
        for (int kk = 0; kk < 4; kk++) {
            v16h nw0, nw1, nah, naw;
            if (kk < 3) {               // prefetch next k-step (4 WMMAs of latency cover)
                const int nkb = (kk + 1) * 32;
                nw0 = loadB(sWih, g0 * HH + hcolw, hi, nkb);
                nw1 = loadB(sWih, g1 * HH + hcolw, hi, nkb);
                nah = loadA(sAh,  nloc, hi, nkb);
                naw = loadA(sAwx, nloc, hi, nkb);
            }
            acc0 = __builtin_amdgcn_wmma_f32_16x16x32_f16(false, cah, false, bhh0[kk], (short)0, acc0, false, false);
            acc1 = __builtin_amdgcn_wmma_f32_16x16x32_f16(false, cah, false, bhh1[kk], (short)0, acc1, false, false);
            acc0 = __builtin_amdgcn_wmma_f32_16x16x32_f16(false, caw, false, cw0,      (short)0, acc0, false, false);
            acc1 = __builtin_amdgcn_wmma_f32_16x16x32_f16(false, caw, false, cw1,      (short)0, acc1, false, false);
            if (kk < 3) { cw0 = nw0; cw1 = nw1; cah = nah; caw = naw; }
        }

        // g,o waves publish their accumulators for the i,f waves
        if (wv >= 8) {
#pragma unroll
            for (int e = 0; e < 8; e++) {
                const int m = e + 8 * hi;
                sG[m * HH + hcolw] = acc0[e];
                sO[m * HH + hcolw] = acc1[e];
            }
        }
        __syncthreads();   // g/o visible; also: all sAh reads done before rewrite below

        // phase 5: LSTM elementwise on i,f waves (acc0=i, acc1=f, sG=g, sO=o)
        if (wv < 8) {
#pragma unroll
            for (int e = 0; e < 8; e++) {
                const int m = e + 8 * hi;
                const float ig = acc0[e], fg = acc1[e];
                const float gg = sG[m * HH + hcolw];
                const float og = sO[m * HH + hcolw];
                const float cold = sC[m * HH + hcolw];
                const float cnew = sigm(fg) * cold + sigm(ig) * tanhf(gg);
                const float hnew = sigm(og) * tanhf(cnew);
                sC[m * HH + hcolw] = cnew;
                sAh[m * DD + hcolw] = (_Float16)hnew;
                out_h[((long)blk * BM + m) * (TM1 * DD) + t * DD + hcolw] = hnew;
            }
        }
        __syncthreads();
    }
}

extern "C" void kernel_launch(void* const* d_in, const int* in_sizes, int n_in,
                              void* d_out, int out_size, void* d_ws, size_t ws_size,
                              hipStream_t stream) {
    (void)in_sizes; (void)n_in; (void)d_ws; (void)ws_size; (void)out_size;
    const float* x     = (const float*)d_in[0];
    const float* Wih   = (const float*)d_in[1];
    const float* Whh   = (const float*)d_in[2];
    const float* bihp  = (const float*)d_in[3];
    const float* bhhp  = (const float*)d_in[4];
    const float* attnw = (const float*)d_in[5];
    const float* attnb = (const float*)d_in[6];
    float* out_w = (float*)d_out;
    float* out_h = (float*)d_out + (long)B_TOT * TM1 * DD;

    dim3 grid(B_TOT / BM);   // 128 blocks
    dim3 block(NTHREADS);    // 16 waves (wave32)
    encoder_lstm_attn_kernel<<<grid, block, 0, stream>>>(x, Wih, Whh, bihp, bhhp,
                                                         attnw, attnb, out_w, out_h);
}
